// VectorQuantizer_75282186764874
// MI455X (gfx1250) — compile-verified, matrix operand order checked by reading, not run on hardware
//
#include <hip/hip_runtime.h>
#include <hip/hip_bf16.h>

// ---------------------------------------------------------------------------
// VQ-VAE quantizer step for MI455X (gfx1250, wave32, WMMA + TDM).
// Heavy op: argmin_k( ||e_k||^2 - 2 * x . e_k ) over K=8192 codes per row,
// N=32768 rows, D=128 -> fused bf16x3 split-precision WMMA GEMM + argmin,
// with embed tiles staged into LDS by the Tensor Data Mover (double buffered)
// and shared across the 8 waves of each workgroup.
// ---------------------------------------------------------------------------

#define VQ_N  32768   // 16*2048 rows
#define VQ_K  8192    // num embeddings
#define VQ_D  128     // embedding dim
#define VQ_KB 64      // codes staged per TDM block
#define VQ_DECAY 0.99f
#define VQ_EPS   1e-05f

typedef __attribute__((ext_vector_type(16))) __bf16 v16bf;
typedef __attribute__((ext_vector_type(8)))  __bf16 v8bf;
typedef __attribute__((ext_vector_type(8)))  float  v8f;
typedef __attribute__((ext_vector_type(4)))  unsigned int v4u;
typedef __attribute__((ext_vector_type(4)))  int    v4i;
typedef __attribute__((ext_vector_type(8)))  int    v8i;

// ---------------------------------------------------------------------------
// 1) f32 -> bf16 (hi, lo) split conversion.  x ~= hi + lo, each bf16.
// ---------------------------------------------------------------------------
__global__ void vq_convert_split(const float* __restrict__ src,
                                 __bf16* __restrict__ hi,
                                 __bf16* __restrict__ lo, int n) {
    int gid = blockIdx.x * blockDim.x + threadIdx.x;
    if (gid < n) {
        float x = src[gid];
        __bf16 h = (__bf16)x;
        float  hf = (float)h;
        __bf16 l = (__bf16)(x - hf);
        hi[gid] = h;
        lo[gid] = l;
    }
}

// ---------------------------------------------------------------------------
// 2) ||e_k||^2 per code, exact f32. One wave32 per code (4 codes / block).
// ---------------------------------------------------------------------------
__global__ void vq_enorm(const float* __restrict__ embed,
                         float* __restrict__ enorm) {
    int lane = threadIdx.x & 31;
    int w    = threadIdx.x >> 5;
    int code = blockIdx.x * 4 + w;
    const float* row = embed + (size_t)code * VQ_D;
    float s = 0.0f;
    #pragma unroll
    for (int i = 0; i < VQ_D / 32; ++i) {
        float v = row[lane + 32 * i];
        s += v * v;
    }
    #pragma unroll
    for (int m = 16; m >= 1; m >>= 1) s += __shfl_xor(s, m, 32);
    if (lane == 0) enorm[code] = s;
}

// ---------------------------------------------------------------------------
// 3) Zero accumulators (counts, sums, n_total) and the loss output slot.
// ---------------------------------------------------------------------------
__global__ void vq_zero(float* __restrict__ counts, float* __restrict__ sums,
                        float* __restrict__ ntot, float* __restrict__ loss_out) {
    int gid = blockIdx.x * blockDim.x + threadIdx.x;
    if (gid < VQ_K * VQ_D) sums[gid] = 0.0f;
    if (gid < VQ_K)        counts[gid] = 0.0f;
    if (gid == 0) { ntot[0] = 0.0f; loss_out[0] = 0.0f; }
}

// ---------------------------------------------------------------------------
// Issue one TDM DMA: copy VQ_KB rows x 256 bytes (bf16 row of D=128) from
// global to LDS.  2D descriptor, data_size = 8 bytes:
//   tensor_dim0 = 32 elems/row, tile = 32 x VQ_KB, stride0 = 32 (contiguous).
// Descriptor fields are workgroup-uniform -> SGPRs, as VIMAGE requires.
// ---------------------------------------------------------------------------
__device__ __forceinline__ void vq_tdm_load(const __bf16* gsrc, unsigned int lds_off) {
    unsigned long long ga = (unsigned long long)(uintptr_t)gsrc;
    v4u g0 = { 1u,                                   // count=1, gather off
               lds_off,                              // lds_addr[31:0]
               (unsigned int)ga,                     // global_addr[31:0]
               ((unsigned int)(ga >> 32) & 0x01FFFFFFu) | 0x80000000u }; // addr[56:32] | type=2
    v8i g1 = { 0x00030000,                           // data_size=3 (8B), mask=0
               0x00200000,                           // tensor_dim0[15:0]=32 << 16
               0x20000000,                           // tensor_dim1[15:0]=8192 << 16
               0x00200000,                           // tile_dim0=32 << 16
               VQ_KB,                                // tile_dim1=VQ_KB, tile_dim2=0
               32,                                   // tensor_dim0_stride=32
               0, 0 };
    v4i z4 = { 0, 0, 0, 0 };
#if defined(__clang_major__) && (__clang_major__ >= 23)
    v8i z8 = { 0, 0, 0, 0, 0, 0, 0, 0 };
    __builtin_amdgcn_tensor_load_to_lds(g0, g1, z4, z4, z8, 0);
#else
    __builtin_amdgcn_tensor_load_to_lds(g0, g1, z4, z4, 0);
#endif
}

// ---------------------------------------------------------------------------
// 4) Fused distance GEMM + argmin.
//    Wave owns 16 rows; A tile (16x128 hi+lo) lives in VGPRs for the whole
//    K sweep.  Embed hi/lo tiles of VQ_KB codes are TDM-staged into LDS,
//    double buffered, shared by all 8 waves.  Per 16-code tile:
//    12x v_wmma_f32_16x16x32_bf16 (bf16x3 split), then running min/argmin;
//    final 16-lane shfl_xor reduction.
//
//    bf16 A 16x32 layout: lane L holds row (L&15); halves 0..7 -> k = 8g+0..7,
//    halves 8..15 -> k = 16+8g+0..7, g = L>>4.  B mirrors with lane&15 = column.
//    C layout: VGPR v, lane L -> row = v + 8g, col = L&15.
// ---------------------------------------------------------------------------
__global__ __launch_bounds__(256)
void vq_dist_argmin(const __bf16* __restrict__ fhi, const __bf16* __restrict__ flo,
                    const __bf16* __restrict__ ehi, const __bf16* __restrict__ elo,
                    const float* __restrict__ enorm,
                    int* __restrict__ idx_out, float* __restrict__ idx_out_f) {
    __shared__ __bf16 s_ehi[2][VQ_KB * VQ_D];   // 2 x 16 KB
    __shared__ __bf16 s_elo[2][VQ_KB * VQ_D];   // 2 x 16 KB

    const int lane = threadIdx.x & 31;
    const int wave = threadIdx.x >> 5;          // 0..7
    const int g    = lane >> 4;                 // 0/1 half-wave
    const int l15  = lane & 15;
    const int row0 = (blockIdx.x * 8 + wave) * 16;

    // ---- load A tile (16 rows x 128 d), hi and lo, into registers ----
    const __bf16* arh = fhi + (size_t)(row0 + l15) * VQ_D;
    const __bf16* arl = flo + (size_t)(row0 + l15) * VQ_D;
    v16bf a_hi[4], a_lo[4];
    #pragma unroll
    for (int c = 0; c < 4; ++c) {
        v8bf h0 = *(const v8bf*)(arh + 32 * c + 8 * g);
        v8bf h1 = *(const v8bf*)(arh + 32 * c + 16 + 8 * g);
        a_hi[c] = __builtin_shufflevector(h0, h1, 0,1,2,3,4,5,6,7,8,9,10,11,12,13,14,15);
        v8bf l0 = *(const v8bf*)(arl + 32 * c + 8 * g);
        v8bf l1 = *(const v8bf*)(arl + 32 * c + 16 + 8 * g);
        a_lo[c] = __builtin_shufflevector(l0, l1, 0,1,2,3,4,5,6,7,8,9,10,11,12,13,14,15);
    }

    float best[8];
    int   bidx[8];
    #pragma unroll
    for (int v = 0; v < 8; ++v) { best[v] = 3.4e38f; bidx[v] = 0; }

    const unsigned int off_ehi0 = (unsigned int)(uintptr_t)&s_ehi[0][0];
    const unsigned int off_ehi1 = (unsigned int)(uintptr_t)&s_ehi[1][0];
    const unsigned int off_elo0 = (unsigned int)(uintptr_t)&s_elo[0][0];
    const unsigned int off_elo1 = (unsigned int)(uintptr_t)&s_elo[1][0];

    // ---- prologue: stage first block into buffer 0 ----
    if (wave == 0) {
        vq_tdm_load(ehi, off_ehi0);
        vq_tdm_load(elo, off_elo0);
    }

    const int kb_count = VQ_K / VQ_KB;          // 128 blocks
    for (int kb = 0; kb < kb_count; ++kb) {
        const int buf = kb & 1;
        if (wave == 0) __builtin_amdgcn_s_wait_tensorcnt(0);
        __syncthreads();                        // buffer `buf` visible to all

        // prefetch next block into the other buffer while we compute
        if (wave == 0 && (kb + 1) < kb_count) {
            const size_t goff = (size_t)(kb + 1) * VQ_KB * VQ_D;
            vq_tdm_load(ehi + goff, buf ? off_ehi0 : off_ehi1);
            vq_tdm_load(elo + goff, buf ? off_elo0 : off_elo1);
        }

        // ---- consume VQ_KB codes = 4 tiles of 16 from LDS ----
        #pragma unroll
        for (int tt = 0; tt < VQ_KB / 16; ++tt) {
            const int cloc = tt * 16 + l15;                 // code within block
            const int code = kb * VQ_KB + cloc;
            const __bf16* brh = &s_ehi[buf][cloc * VQ_D];
            const __bf16* brl = &s_elo[buf][cloc * VQ_D];
            v8f acc = {};
            #pragma unroll
            for (int c = 0; c < 4; ++c) {
                v8bf bh0 = *(const v8bf*)(brh + 32 * c + 8 * g);
                v8bf bh1 = *(const v8bf*)(brh + 32 * c + 16 + 8 * g);
                v16bf bh = __builtin_shufflevector(bh0, bh1, 0,1,2,3,4,5,6,7,8,9,10,11,12,13,14,15);
                v8bf bl0 = *(const v8bf*)(brl + 32 * c + 8 * g);
                v8bf bl1 = *(const v8bf*)(brl + 32 * c + 16 + 8 * g);
                v16bf bl = __builtin_shufflevector(bl0, bl1, 0,1,2,3,4,5,6,7,8,9,10,11,12,13,14,15);
                // bf16x3 split product: hi*hi + hi*lo + lo*hi, f32 accumulate
                acc = __builtin_amdgcn_wmma_f32_16x16x32_bf16(false, a_hi[c], false, bh,
                                                              (short)0, acc, false, false);
                acc = __builtin_amdgcn_wmma_f32_16x16x32_bf16(false, a_hi[c], false, bl,
                                                              (short)0, acc, false, false);
                acc = __builtin_amdgcn_wmma_f32_16x16x32_bf16(false, a_lo[c], false, bh,
                                                              (short)0, acc, false, false);
            }
            const float en = enorm[code];
            #pragma unroll
            for (int v = 0; v < 8; ++v) {
                float d = en - 2.0f * acc[v];   // ||e||^2 - 2 x.e (row const dropped)
                if (d < best[v]) { best[v] = d; bidx[v] = code; }
            }
        }
        __syncthreads();                        // all waves done with `buf`
    }

    // ---- reduce across the 16 columns held by the half-wave ----
    #pragma unroll
    for (int m = 1; m <= 8; m <<= 1) {
        #pragma unroll
        for (int v = 0; v < 8; ++v) {
            float ob = __shfl_xor(best[v], m, 32);
            int   oi = __shfl_xor(bidx[v], m, 32);
            if (ob < best[v] || (ob == best[v] && oi < bidx[v])) {
                best[v] = ob; bidx[v] = oi;
            }
        }
    }
    if (l15 == 0) {
        #pragma unroll
        for (int v = 0; v < 8; ++v) {
            int r = row0 + v + 8 * g;
            idx_out[r]   = bidx[v];
            idx_out_f[r] = (float)bidx[v];
        }
    }
}

// ---------------------------------------------------------------------------
// 5) Gather z_q (straight-through output equals z_q numerically), commitment
//    loss partial, and atomic segment sums (counts, per-dim sums).
//    One block = one row (128 threads).
// ---------------------------------------------------------------------------
__global__ void vq_gather_stats(const float* __restrict__ z,
                                const float* __restrict__ embed,
                                const int* __restrict__ idx,
                                float* __restrict__ zq_out,
                                float* __restrict__ loss_out,
                                float* __restrict__ counts,
                                float* __restrict__ sums) {
    const int n = blockIdx.x;
    const int d = threadIdx.x;
    const int code = idx[n];
    const float zv = z[(size_t)n * VQ_D + d];
    const float ev = embed[(size_t)code * VQ_D + d];
    zq_out[(size_t)n * VQ_D + d] = ev;

    float diff = zv - ev;
    float s = diff * diff;
    #pragma unroll
    for (int m = 16; m >= 1; m >>= 1) s += __shfl_xor(s, m, 32);
    __shared__ float red[4];
    const int lane = threadIdx.x & 31, w = threadIdx.x >> 5;
    if (lane == 0) red[w] = s;
    __syncthreads();
    if (threadIdx.x == 0) {
        float tot = red[0] + red[1] + red[2] + red[3];
        atomicAdd(loss_out, tot * (1.0f / ((float)VQ_N * (float)VQ_D)));
        atomicAdd(&counts[code], 1.0f);
    }
    atomicAdd(&sums[(size_t)code * VQ_D + d], zv);
}

// ---------------------------------------------------------------------------
// 6) EMA update: new_size, new_sum, and n = sum(new_size).
// ---------------------------------------------------------------------------
__global__ void vq_ema(const float* __restrict__ cluster_size,
                       const float* __restrict__ embed_avg,
                       const float* __restrict__ counts,
                       const float* __restrict__ sums,
                       float* __restrict__ new_size,
                       float* __restrict__ new_sum,
                       float* __restrict__ ntot) {
    int gid = blockIdx.x * blockDim.x + threadIdx.x;
    if (gid < VQ_K * VQ_D)
        new_sum[gid] = VQ_DECAY * embed_avg[gid] + (1.0f - VQ_DECAY) * sums[gid];
    if (gid < VQ_K) {
        float ns = VQ_DECAY * cluster_size[gid] + (1.0f - VQ_DECAY) * counts[gid];
        new_size[gid] = ns;
        atomicAdd(ntot, ns);
    }
}

// ---------------------------------------------------------------------------
// 7) Laplace smoothing + normalized codebook.
// ---------------------------------------------------------------------------
__global__ void vq_final(const float* __restrict__ new_size,
                         const float* __restrict__ new_sum,
                         const float* __restrict__ ntot,
                         float* __restrict__ new_embed) {
    int gid = blockIdx.x * blockDim.x + threadIdx.x;
    if (gid < VQ_K * VQ_D) {
        float n  = *ntot;
        float ns = new_size[gid >> 7];   // gid / 128
        float smoothed = (ns + VQ_EPS) / (n + (float)VQ_K * VQ_EPS) * n;
        new_embed[gid] = new_sum[gid] / smoothed;
    }
}

// ---------------------------------------------------------------------------
// Launcher
// ---------------------------------------------------------------------------
extern "C" void kernel_launch(void* const* d_in, const int* in_sizes, int n_in,
                              void* d_out, int out_size, void* d_ws, size_t ws_size,
                              hipStream_t stream) {
    const float* z            = (const float*)d_in[0];  // [N, D]
    const float* embed        = (const float*)d_in[1];  // [K, D]
    const float* cluster_size = (const float*)d_in[2];  // [K]
    const float* embed_avg    = (const float*)d_in[3];  // [K, D]

    // ---- d_out layout (floats, reference return order) ----
    float* out      = (float*)d_out;
    float* out_zq   = out;                              // N*D
    float* out_loss = out + (size_t)VQ_N * VQ_D;        // 1
    float* out_idx  = out_loss + 1;                     // N
    float* out_nsz  = out_idx + VQ_N;                   // K
    float* out_nsm  = out_nsz + VQ_K;                   // K*D
    float* out_nem  = out_nsm + (size_t)VQ_K * VQ_D;    // K*D

    // ---- workspace layout ----
    char* ws = (char*)d_ws;
    size_t o = 0;
    __bf16* fhi = (__bf16*)(ws + o); o += (size_t)VQ_N * VQ_D * 2;
    __bf16* flo = (__bf16*)(ws + o); o += (size_t)VQ_N * VQ_D * 2;
    __bf16* ehi = (__bf16*)(ws + o); o += (size_t)VQ_K * VQ_D * 2;
    __bf16* elo = (__bf16*)(ws + o); o += (size_t)VQ_K * VQ_D * 2;
    float*  enorm  = (float*)(ws + o); o += (size_t)VQ_K * 4;
    int*    idx    = (int*)  (ws + o); o += (size_t)VQ_N * 4;
    float*  counts = (float*)(ws + o); o += (size_t)VQ_K * 4;
    float*  sums   = (float*)(ws + o); o += (size_t)VQ_K * VQ_D * 4;
    float*  ntot   = (float*)(ws + o); o += 256;

    const int nd = VQ_N * VQ_D;   // 4194304
    const int kd = VQ_K * VQ_D;   // 1048576

    // 1) split-precision conversions
    vq_convert_split<<<(nd + 255) / 256, 256, 0, stream>>>(z, fhi, flo, nd);
    vq_convert_split<<<(kd + 255) / 256, 256, 0, stream>>>(embed, ehi, elo, kd);
    // 2) code norms
    vq_enorm<<<VQ_K / 4, 128, 0, stream>>>(embed, enorm);
    // 3) zero accumulators + loss slot
    vq_zero<<<(kd + 255) / 256, 256, 0, stream>>>(counts, sums, ntot, out_loss);
    // 4) fused WMMA distance + argmin (256 WGs x 8 waves x 16 rows = 32768)
    vq_dist_argmin<<<VQ_N / 128, 256, 0, stream>>>(fhi, flo, ehi, elo, enorm,
                                                   idx, out_idx);
    // 5) gather z_q, loss, segment sums
    vq_gather_stats<<<VQ_N, VQ_D, 0, stream>>>(z, embed, idx, out_zq, out_loss,
                                               counts, sums);
    // 6) EMA
    vq_ema<<<(kd + 255) / 256, 256, 0, stream>>>(cluster_size, embed_avg, counts,
                                                 sums, out_nsz, out_nsm, ntot);
    // 7) Laplace-smoothed codebook
    vq_final<<<(kd + 255) / 256, 256, 0, stream>>>(out_nsz, out_nsm, ntot, out_nem);
}
